// VolterraLayer2D_12360915878397
// MI455X (gfx1250) — matrix operand order repaired and checked by reading, not compiled, not run on hardware
//
#include <hip/hip_runtime.h>

typedef __attribute__((ext_vector_type(16))) _Float16 v16h;
typedef __attribute__((ext_vector_type(8)))  _Float16 v8h;
typedef __attribute__((ext_vector_type(8)))  float    v8f;
typedef __attribute__((ext_vector_type(4)))  float    v4f;
typedef __attribute__((ext_vector_type(4)))  unsigned int v4u;
typedef __attribute__((ext_vector_type(8)))  int      v8i;
typedef __attribute__((ext_vector_type(4)))  int      v4i;

static constexpr int kC   = 64;     // CI == CO
static constexpr int kH   = 256;
static constexpr int kW   = 256;
static constexpr int kB   = 8;
static constexpr int kTH  = 8;      // pixel tile rows per workgroup
static constexpr int kTW  = 16;     // pixel tile cols per workgroup
static constexpr int kXR  = 10;     // tile rows + halo
static constexpr int kXC  = 18;     // tile cols + halo
static constexpr int kPS  = 72;     // halfs per pixel in LDS (144B stride -> conflict-free b128)
static constexpr int kWTS = 648;    // halfs per CO row of packed weights (1296B stride)
static constexpr int kWTBYTES = kC * kWTS * 2;   // 82944
static constexpr int kWT8 = kWTBYTES / 8;        // 10368 8-byte units for the TDM descriptor

__device__ __forceinline__ v16h cat8(v8h a, v8h b) {
  return __builtin_shufflevector(a, b, 0,1,2,3,4,5,6,7,8,9,10,11,12,13,14,15);
}

// ---------------------------------------------------------------------------
// Prep: pack B matrix Wt[64][648] (f16) = [9 conv taps x 64 CI | M = sum_q Wb.Wa]
// and combined bias = linear_b + sum_q (Wb.ab + bb).
// ---------------------------------------------------------------------------
__global__ __launch_bounds__(256) void volterra_prep(
    const float* __restrict__ lw,  const float* __restrict__ lb,
    const float* __restrict__ w2aw, const float* __restrict__ w2ab,
    const float* __restrict__ w2bw, const float* __restrict__ w2bb,
    _Float16* __restrict__ Wt, float* __restrict__ biasAll)
{
  const int tid = threadIdx.x;
  // conv taps: Wt[o][tap*64+ci] = lw[o][ci][ky][kx], tap = ky*3+kx
  for (int e = tid; e < kC * 576; e += 256) {
    int o = e / 576, k = e - o * 576;
    int tap = k >> 6, ci = k & 63;
    Wt[o * kWTS + k] = (_Float16)lw[(o * kC + ci) * 9 + tap];
  }
  // M[o][i] = sum_q sum_c w2bw[q][o][c] * w2aw[q][c][i]
  for (int e = tid; e < kC * kC; e += 256) {
    int o = e >> 6, i = e & 63;
    float m = 0.f;
    for (int q = 0; q < 4; ++q)
      for (int c = 0; c < kC; ++c)
        m += w2bw[(q * kC + o) * kC + c] * w2aw[(q * kC + c) * kC + i];
    Wt[o * kWTS + 576 + i] = (_Float16)m;
  }
  if (tid < kC) {
    int o = tid;
    float bb = lb[o];
    for (int q = 0; q < 4; ++q) {
      float t = 0.f;
      for (int c = 0; c < kC; ++c)
        t += w2bw[(q * kC + o) * kC + c] * w2ab[q * kC + c];
      bb += t + w2bb[q * kC + o];
    }
    biasAll[o] = bb;
  }
}

// ---------------------------------------------------------------------------
// Main fused kernel: one 8x16 pixel tile x all 64 CO per workgroup.
// Weights arrive via Tensor Data Mover (async, overlapped with phases 1-2).
// ---------------------------------------------------------------------------
__global__ __launch_bounds__(128) void volterra_main(
    const float* __restrict__ x, const _Float16* __restrict__ Wt,
    const float* __restrict__ biasAll, float* __restrict__ out)
{
  __shared__ _Float16 s_xh[kXR * kXC * kPS];     // 25920 B: f16 x, OOB halo zeroed (conv A)
  __shared__ _Float16 s_xc[kXR * kXC * kPS];     // 25920 B: f16 x, circular halo (for Sn)
  __shared__ _Float16 s_snh[kTH * kTW * kPS];    // 18432 B: f16 Sn tile
  __shared__ float    s_bias[kC];                //   256 B
  __shared__ __align__(16) _Float16 s_wt[kC * kWTS];   // 82944 B: packed weights via TDM

  const int tid = threadIdx.x;
  const int lane = tid & 31, wv = tid >> 5;
  const int tw = blockIdx.x, th = blockIdx.y, b = blockIdx.z;
  const int h0 = th * kTH, w0 = tw * kTW;

  // ---- phase 0: kick off async TDM load of the 82944B weight block into LDS.
  // D# group0: count=1, lds_addr, 57-bit global_addr, type=2 ("image").
  // D# group1: data_size=8B (3), tensor_dim0=tile_dim0=10368, tensor_dim1=1,
  //            tensor_dim0_stride=10368. Groups 2/3 (and trailing group) zero.
  if (wv == 0) {
    unsigned lds_off = (unsigned)(unsigned long long)(uintptr_t)(void*)s_wt;
    unsigned long long ga = (unsigned long long)(uintptr_t)Wt;
    v4u g0 = { 1u,                                   // count=1, is_restore=0, gather=0
               lds_off,                              // lds_addr (bytes)
               (unsigned)(ga & 0xffffffffu),         // global_addr[31:0]
               (unsigned)((ga >> 32) & 0x01ffffffu) | 0x80000000u }; // addr[56:32] | type=2
    v8i g1 = { 0x00030000,                           // wg_mask=0, data_size=3 (8B)
               (int)(kWT8 << 16),                    // tensor_dim0[15:0] in [31:16]
               0x00010000,                           // tensor_dim1=1
               (int)(kWT8 << 16),                    // tile_dim0 in [31:16]
               0,                                    // tile_dim1=0, tile_dim2=0
               kWT8,                                 // tensor_dim0_stride[31:0]
               0, 0 };
    v4i g2 = { 0, 0, 0, 0 };
    v4i g3 = { 0, 0, 0, 0 };
    v8i g4 = { 0, 0, 0, 0, 0, 0, 0, 0 };
    __builtin_amdgcn_tensor_load_to_lds(g0, g1, g2, g3, g4, 0);
  }

  // ---- phase 1: load x tile (circular wrap) -> circular f16 + zero-masked f16
  for (int idx = tid; idx < kC * kXR * kXC; idx += 128) {
    int ci  = idx / (kXR * kXC);
    int rem = idx - ci * (kXR * kXC);
    int r = rem / kXC, c = rem - r * kXC;
    int hg = h0 - 1 + r, wg = w0 - 1 + c;
    int hcirc = hg & (kH - 1), wcirc = wg & (kW - 1);
    float v = x[(((size_t)b * kC + ci) * kH + hcirc) * kW + wcirc];
    _Float16 hv = (_Float16)v;
    int p = r * kXC + c;
    s_xc[p * kPS + ci] = hv;
    bool in = (hg >= 0) && (hg < kH) && (wg >= 0) && (wg < kW);
    s_xh[p * kPS + ci] = in ? hv : (_Float16)0.0f;
  }
  if (tid < kC) s_bias[tid] = biasAll[tid];
  __syncthreads();

  // ---- phase 2: Sn = 0.5*((sum x)^2 + sum x^2)/45 over circular 3x3, -> f16
  for (int idx = tid; idx < kTH * kTW * kC; idx += 128) {
    int ci = idx & 63;
    int p  = idx >> 6;
    int r = p >> 4, c = p & 15;
    float s1 = 0.f, s2 = 0.f;
    #pragma unroll
    for (int dy = 0; dy < 3; ++dy)
      #pragma unroll
      for (int dx = 0; dx < 3; ++dx) {
        float v = (float)s_xc[((r + dy) * kXC + (c + dx)) * kPS + ci];
        s1 += v; s2 += v * v;
      }
    float sn = 0.5f * (s1 * s1 + s2) * (1.0f / 45.0f);
    s_snh[p * kPS + ci] = (_Float16)sn;
  }

  // weights must have landed before the GEMM reads them
  if (wv == 0) __builtin_amdgcn_s_wait_tensorcnt(0);
  __syncthreads();

  // ---- phase 3: WMMA implicit GEMM
  const int l16 = lane & 15, lh = lane >> 4;
  const int r0 = wv * 2;
  const _Float16* wl = s_wt;

  v8f acc[2][4];
  #pragma unroll
  for (int m = 0; m < 2; ++m)
    #pragma unroll
    for (int n = 0; n < 4; ++n)
      #pragma unroll
      for (int g = 0; g < 8; ++g) acc[m][n][g] = 0.f;

  // conv taps: K-steps 0..17
  for (int t = 0; t < 9; ++t) {
    const int dy = t / 3, dx = t - 3 * (t / 3);
    #pragma unroll
    for (int hk = 0; hk < 2; ++hk) {
      const int k0 = (t * 2 + hk) * 32;
      const int ci0 = hk * 32;
      v16h A[2];
      #pragma unroll
      for (int m = 0; m < 2; ++m) {
        int aoff = ((r0 + m + dy) * kXC + (l16 + dx)) * kPS + ci0 + lh * 8;
        v8h a0 = *(const v8h*)(s_xh + aoff);
        v8h a1 = *(const v8h*)(s_xh + aoff + 16);
        A[m] = cat8(a0, a1);
      }
      #pragma unroll
      for (int n = 0; n < 4; ++n) {
        int boff = (n * 16 + l16) * kWTS + k0 + lh * 8;
        v8h b0 = *(const v8h*)(wl + boff);
        v8h b1 = *(const v8h*)(wl + boff + 16);
        v16h Bv = cat8(b0, b1);
        #pragma unroll
        for (int m = 0; m < 2; ++m)
          acc[m][n] = __builtin_amdgcn_wmma_f32_16x16x32_f16(
              false, A[m], false, Bv, (short)0, acc[m][n], false, false);
      }
    }
  }

  // quadratic (Sn x M): K-steps 18..19
  #pragma unroll
  for (int hk = 0; hk < 2; ++hk) {
    const int k0 = (18 + hk) * 32;
    const int ci0 = hk * 32;
    v16h A[2];
    #pragma unroll
    for (int m = 0; m < 2; ++m) {
      int p = (r0 + m) * kTW + l16;
      int aoff = p * kPS + ci0 + lh * 8;
      v8h a0 = *(const v8h*)(s_snh + aoff);
      v8h a1 = *(const v8h*)(s_snh + aoff + 16);
      A[m] = cat8(a0, a1);
    }
    #pragma unroll
    for (int n = 0; n < 4; ++n) {
      int boff = (n * 16 + l16) * kWTS + k0 + lh * 8;
      v8h b0 = *(const v8h*)(wl + boff);
      v8h b1 = *(const v8h*)(wl + boff + 16);
      v16h Bv = cat8(b0, b1);
      #pragma unroll
      for (int m = 0; m < 2; ++m)
        acc[m][n] = __builtin_amdgcn_wmma_f32_16x16x32_f16(
            false, A[m], false, Bv, (short)0, acc[m][n], false, false);
    }
  }

  // ---- epilogue: + bias, store NCHW f32. Lane holds co = n*16+l16, 8 consecutive w.
  #pragma unroll
  for (int m = 0; m < 2; ++m) {
    const int hrow = h0 + r0 + m;
    #pragma unroll
    for (int n = 0; n < 4; ++n) {
      const int co = n * 16 + l16;
      const float bc = s_bias[co];
      float* op = out + (((size_t)b * kC + co) * kH + hrow) * kW + w0 + lh * 8;
      v4f lo = { acc[m][n][0] + bc, acc[m][n][1] + bc, acc[m][n][2] + bc, acc[m][n][3] + bc };
      v4f hi = { acc[m][n][4] + bc, acc[m][n][5] + bc, acc[m][n][6] + bc, acc[m][n][7] + bc };
      *(v4f*)op       = lo;
      *(v4f*)(op + 4) = hi;
    }
  }
}

extern "C" void kernel_launch(void* const* d_in, const int* in_sizes, int n_in,
                              void* d_out, int out_size, void* d_ws, size_t ws_size,
                              hipStream_t stream) {
  const float* x    = (const float*)d_in[0];
  const float* lw   = (const float*)d_in[1];
  const float* lb   = (const float*)d_in[2];
  const float* w2aw = (const float*)d_in[3];
  const float* w2ab = (const float*)d_in[4];
  const float* w2bw = (const float*)d_in[5];
  const float* w2bb = (const float*)d_in[6];
  float* out = (float*)d_out;

  _Float16* Wt   = (_Float16*)d_ws;
  float* biasAll = (float*)((char*)d_ws + kWTBYTES);

  volterra_prep<<<1, 256, 0, stream>>>(lw, lb, w2aw, w2ab, w2bw, w2bb, Wt, biasAll);

  dim3 grid(kW / kTW, kH / kTH, kB);   // 16 x 32 x 8 = 4096 workgroups
  volterra_main<<<grid, 128, 0, stream>>>(x, Wt, biasAll, out);
}
